// PertAdapterNew_24575802868165
// MI455X (gfx1250) — matrix operand order repaired and checked by reading, not compiled, not run on hardware
//
#include <hip/hip_runtime.h>
#include <hip/hip_bf16.h>

// ---------------------------------------------------------------------------
// Transformer encoder layer for MI455X (gfx1250, wave32, WMMA + TDM).
// B=1, S=2048, D=768, H=12, HD=64.
// GEMMs: 128x128 block tile, K-step 64, double-buffered LDS fed by the
// Tensor Data Mover (tensor_load_to_lds), overlapped with WMMA via
// s_wait_tensorcnt.  Attention: flash-style with online softmax, WMMA for
// both Q*K^T and P*V.  Workspace requirement: ~46 MB.
// ---------------------------------------------------------------------------

typedef __bf16 bf16_t;
typedef __attribute__((ext_vector_type(16))) __bf16 v16bf;
typedef __attribute__((ext_vector_type(8)))  float  v8f;
typedef __attribute__((ext_vector_type(4)))  unsigned int v4u;
typedef __attribute__((ext_vector_type(8)))  int v8i;
typedef __attribute__((ext_vector_type(4)))  int v4i;

union BF16Frag {
    v16bf v;
    uint4 q[2];
};

static __device__ __forceinline__ bf16_t f2bf(float x) { return (bf16_t)x; }

#define WMMA_BF16(A, B, C) \
    __builtin_amdgcn_wmma_f32_16x16x32_bf16(false, (A), false, (B), (short)0, (C), false, false)

#if defined(__has_builtin)
#if __has_builtin(__builtin_amdgcn_tensor_load_to_lds) && __has_builtin(__builtin_amdgcn_s_wait_tensorcnt)
#define HAVE_TDM 1
#endif
#endif
#ifndef HAVE_TDM
#define HAVE_TDM 0
#endif

#if HAVE_TDM
// Issue a 2-D TDM load: tile_d1 rows x tile_d0 bf16 elems, row stride
// stride0 (elems) in global, packed contiguously (row-major) into LDS.
// D# bit layout per CDNA5 ISA 8.3/8.4 (group0 128b, group1 256b).
static __device__ __forceinline__ void tdm_load_2d(unsigned lds_byte_off,
                                                   const void* gptr,
                                                   unsigned tile_d0, unsigned tile_d1,
                                                   unsigned long long stride0) {
    unsigned long long ga = (unsigned long long)(size_t)gptr;
    v4u g0;
    g0.x = 1u;                                            // count=1, user desc
    g0.y = lds_byte_off;                                  // lds_addr
    g0.z = (unsigned)(ga & 0xffffffffu);                  // global_addr[31:0]
    g0.w = (unsigned)((ga >> 32) & 0x01ffffffu) | (2u << 30);  // addr[56:32] | type=2
    v8i g1;
    g1[0] = (int)(1u << 16);                              // wg_mask=0, data_size=1 (2B)
    g1[1] = (int)((tile_d0 & 0xffffu) << 16);             // tensor_dim0 = tile_d0 (exact)
    g1[2] = (int)((tile_d0 >> 16) | ((tile_d1 & 0xffffu) << 16));  // tensor_dim1 = tile_d1
    g1[3] = (int)((tile_d1 >> 16) | (tile_d0 << 16));     // tile_dim0
    g1[4] = (int)(tile_d1 & 0xffffu);                     // tile_dim1, tile_dim2=0
    g1[5] = (int)(stride0 & 0xffffffffu);                 // tensor_dim0_stride lo
    g1[6] = (int)((stride0 >> 32) & 0xffffu);             // stride hi, dim1_stride=0
    g1[7] = 0;
    v4i z4 = {0, 0, 0, 0};
#if __clang_major__ >= 23
    v8i z8 = {0, 0, 0, 0, 0, 0, 0, 0};
    __builtin_amdgcn_tensor_load_to_lds(g0, g1, z4, z4, z8, 0);
#else
    __builtin_amdgcn_tensor_load_to_lds(g0, g1, z4, z4, 0);
#endif
}
#endif

// --------------------------- f32 -> bf16 convert ---------------------------
__global__ void k_f32_to_bf16(const float* __restrict__ in, bf16_t* __restrict__ out, int n) {
    int i = blockIdx.x * blockDim.x + threadIdx.x;
    int stride = gridDim.x * blockDim.x;
    for (; i < n; i += stride) out[i] = f2bf(in[i]);
}

// --------------------------- fused add + LayerNorm --------------------------
__global__ __launch_bounds__(256)
void k_add_ln(const float* __restrict__ a, const float* __restrict__ b,
              const float* __restrict__ gamma, const float* __restrict__ beta,
              float* __restrict__ outf, bf16_t* __restrict__ outb, int D) {
    const int row = blockIdx.x;
    const float* pa = a + (size_t)row * D;
    const float* pb = b + (size_t)row * D;
    float v[3];
    float s = 0.f;
#pragma unroll
    for (int i = 0; i < 3; ++i) {
        int c = threadIdx.x + i * 256;
        v[i] = pa[c] + pb[c];
        s += v[i];
    }
    __shared__ float red[8];
    for (int off = 16; off; off >>= 1) s += __shfl_xor(s, off, 32);
    const int w = threadIdx.x >> 5;
    if ((threadIdx.x & 31) == 0) red[w] = s;
    __syncthreads();
    float tot = 0.f;
#pragma unroll
    for (int i = 0; i < 8; ++i) tot += red[i];
    const float mu = tot / (float)D;
    float vp = 0.f;
#pragma unroll
    for (int i = 0; i < 3; ++i) { float d = v[i] - mu; vp += d * d; }
    for (int off = 16; off; off >>= 1) vp += __shfl_xor(vp, off, 32);
    __syncthreads();
    if ((threadIdx.x & 31) == 0) red[w] = vp;
    __syncthreads();
    float vt = 0.f;
#pragma unroll
    for (int i = 0; i < 8; ++i) vt += red[i];
    const float rstd = rsqrtf(vt / (float)D + 1e-5f);
#pragma unroll
    for (int i = 0; i < 3; ++i) {
        int c = threadIdx.x + i * 256;
        float y = (v[i] - mu) * rstd * gamma[c] + beta[c];
        if (outf) outf[(size_t)row * D + c] = y;
        if (outb) outb[(size_t)row * D + c] = f2bf(y);
    }
}

// --------------------------- WMMA GEMM: C = A * W^T + bias ------------------
// A: [M,K] bf16.  W: [N,K] bf16.  Block tile 128x128, K-step 64, 8 waves,
// wave tile 32x64.  Double-buffered LDS, TDM-fed (wave 0 issues DMA for the
// next K-tile while all waves run WMMA on the current one).
// M,N multiples of 128; K multiple of 64.
__global__ __launch_bounds__(256)
void k_gemm_wmma(const bf16_t* __restrict__ A, const bf16_t* __restrict__ W,
                 const float* __restrict__ bias,
                 float* __restrict__ Cf, bf16_t* __restrict__ Cb,
                 int M, int N, int K, int relu) {
    __shared__ bf16_t Atile[2][128][64];
    __shared__ bf16_t Btile[2][128][64];
    const int tid  = threadIdx.x;
    const int lane = tid & 31;
    const int w    = tid >> 5;
    const int g    = lane >> 4;       // lane half (0/1)
    const int ln   = lane & 15;
    const int wm   = (w & 3) * 32;    // wave row offset (0..96)
    const int wn   = (w >> 2) * 64;   // wave col offset (0/64)
    const int bm   = blockIdx.y * 128;
    const int bn   = blockIdx.x * 128;

    v8f acc[2][4] = {};
    const int nIter = K >> 6;

#if HAVE_TDM
    if (w == 0) {
        tdm_load_2d((unsigned)(size_t)&Atile[0][0][0], &A[(size_t)bm * K], 64, 128, (unsigned)K);
        tdm_load_2d((unsigned)(size_t)&Btile[0][0][0], &W[(size_t)bn * K], 64, 128, (unsigned)K);
    }
#endif

    for (int it = 0; it < nIter; ++it) {
        const int cur = it & 1;
        const int kk  = it << 6;
#if HAVE_TDM
        if (w == 0) {
            if (it + 1 < nIter) {
                tdm_load_2d((unsigned)(size_t)&Atile[cur ^ 1][0][0],
                            &A[(size_t)bm * K + kk + 64], 64, 128, (unsigned)K);
                tdm_load_2d((unsigned)(size_t)&Btile[cur ^ 1][0][0],
                            &W[(size_t)bn * K + kk + 64], 64, 128, (unsigned)K);
                __builtin_amdgcn_s_wait_tensorcnt(2);  // current pair complete
            } else {
                __builtin_amdgcn_s_wait_tensorcnt(0);
            }
        }
        __syncthreads();  // broadcast tile readiness
#else
        // fallback: cooperative staging (4x b128 per thread per tile)
#pragma unroll
        for (int i = 0; i < 4; ++i) {
            int idx = tid + i * 256;
            int r = idx >> 3, ch = (idx & 7) * 8;
            *(uint4*)&Atile[cur][r][ch] = *(const uint4*)&A[(size_t)(bm + r) * K + kk + ch];
            *(uint4*)&Btile[cur][r][ch] = *(const uint4*)&W[(size_t)(bn + r) * K + kk + ch];
        }
        __syncthreads();
#endif

#pragma unroll
        for (int ks = 0; ks < 2; ++ks) {
            BF16Frag af[2], bf[4];
#pragma unroll
            for (int fa = 0; fa < 2; ++fa) {
                const bf16_t* p = &Atile[cur][wm + fa * 16 + ln][ks * 32];
                af[fa].q[0] = *(const uint4*)(p + 8 * g);
                af[fa].q[1] = *(const uint4*)(p + 16 + 8 * g);
            }
#pragma unroll
            for (int fb = 0; fb < 4; ++fb) {
                const bf16_t* p = &Btile[cur][wn + fb * 16 + ln][ks * 32 + 8 * g];
                bf[fb].q[0] = *(const uint4*)p;
                bf[fb].q[1] = *(const uint4*)(p + 16);
            }
#pragma unroll
            for (int fa = 0; fa < 2; ++fa)
#pragma unroll
                for (int fb = 0; fb < 4; ++fb)
                    acc[fa][fb] = WMMA_BF16(af[fa].v, bf[fb].v, acc[fa][fb]);
        }
        __syncthreads();  // reads done before next DMA overwrites this buffer
    }

    // epilogue: C layout -> row = j + 8g, col = ln per 16x16 tile
#pragma unroll
    for (int fa = 0; fa < 2; ++fa)
#pragma unroll
        for (int fb = 0; fb < 4; ++fb) {
            const int col = bn + wn + fb * 16 + ln;
            const float bv = bias ? bias[col] : 0.f;
#pragma unroll
            for (int j = 0; j < 8; ++j) {
                const int row = bm + wm + fa * 16 + j + 8 * g;
                float v = acc[fa][fb][j] + bv;
                if (relu) v = fmaxf(v, 0.f);
                if (Cf) Cf[(size_t)row * N + col] = v;
                if (Cb) Cb[(size_t)row * N + col] = f2bf(v);
            }
        }
}

// --------------------------- masked flash attention --------------------------
__global__ __launch_bounds__(256)
void k_attention(const bf16_t* __restrict__ qkv, const int* __restrict__ adj,
                 bf16_t* __restrict__ o, int S, int D, int H) {
    __shared__ bf16_t Kt[32][64];       // [key][dim]
    __shared__ bf16_t Vt[64][32];       // [dim][key] (transposed)
    __shared__ bf16_t Pw[8][16][32];    // per-wave probs, A-layout staging

    const int tid  = threadIdx.x;
    const int lane = tid & 31;
    const int w    = tid >> 5;
    const int g    = lane >> 4;
    const int ln   = lane & 15;
    const int h    = blockIdx.y;
    const int q0   = blockIdx.x * 128 + w * 16;
    const int ld   = 3 * D;  // 2304

    BF16Frag qa[2];
#pragma unroll
    for (int s2 = 0; s2 < 2; ++s2) {
        const bf16_t* p = qkv + (size_t)(q0 + ln) * ld + h * 64 + s2 * 32;
        qa[s2].q[0] = *(const uint4*)(p + 8 * g);
        qa[s2].q[1] = *(const uint4*)(p + 16 + 8 * g);
    }

    v8f accO[4] = {};
    float m_run[8], l_run[8];
#pragma unroll
    for (int j = 0; j < 8; ++j) { m_run[j] = -1e30f; l_run[j] = 0.f; }

    for (int kb = 0; kb < S; kb += 32) {
        __syncthreads();
        {
            const int key = tid >> 3, ch = (tid & 7) * 8;
            const bf16_t* ksrc = qkv + (size_t)(kb + key) * ld + D + h * 64 + ch;
            *(uint4*)&Kt[key][ch] = *(const uint4*)ksrc;
            const bf16_t* vsrc = qkv + (size_t)(kb + key) * ld + 2 * D + h * 64 + ch;
            uint4 vv = *(const uint4*)vsrc;
            const bf16_t* ve = (const bf16_t*)&vv;
#pragma unroll
            for (int e = 0; e < 8; ++e) Vt[ch + e][key] = ve[e];
        }
        __syncthreads();

        v8f sf[2];
#pragma unroll
        for (int nt = 0; nt < 2; ++nt) {
            v8f c = {};
#pragma unroll
            for (int s2 = 0; s2 < 2; ++s2) {
                BF16Frag b;
                const bf16_t* p = &Kt[nt * 16 + ln][s2 * 32 + 8 * g];
                b.q[0] = *(const uint4*)p;
                b.q[1] = *(const uint4*)(p + 16);
                c = WMMA_BF16(qa[s2].v, b.v, c);
            }
            const int col = kb + nt * 16 + ln;
#pragma unroll
            for (int j = 0; j < 8; ++j) {
                const int qrow = q0 + j + 8 * g;
                float sv = c[j] * 0.125f;  // 1/sqrt(64)
                if (adj[(size_t)qrow * S + col] == 0) sv = -1e30f;
                c[j] = sv;
            }
            sf[nt] = c;
        }

        float m_new[8];
#pragma unroll
        for (int j = 0; j < 8; ++j) {
            float mx = fmaxf(sf[0][j], sf[1][j]);
            for (int off = 8; off; off >>= 1) mx = fmaxf(mx, __shfl_xor(mx, off, 32));
            m_new[j] = fmaxf(m_run[j], mx);
        }
#pragma unroll
        for (int nt = 0; nt < 2; ++nt)
#pragma unroll
            for (int j = 0; j < 8; ++j)
                sf[nt][j] = __expf(sf[nt][j] - m_new[j]);
#pragma unroll
        for (int j = 0; j < 8; ++j) {
            float ls = sf[0][j] + sf[1][j];
            for (int off = 8; off; off >>= 1) ls += __shfl_xor(ls, off, 32);
            const float scale = __expf(m_run[j] - m_new[j]);
            l_run[j] = l_run[j] * scale + ls;
            m_run[j] = m_new[j];
#pragma unroll
            for (int t = 0; t < 4; ++t) accO[t][j] *= scale;
        }

#pragma unroll
        for (int nt = 0; nt < 2; ++nt)
#pragma unroll
            for (int j = 0; j < 8; ++j)
                Pw[w][j + 8 * g][nt * 16 + ln] = f2bf(sf[nt][j]);
        asm volatile("s_wait_dscnt 0" ::: "memory");  // LDS RAW within wave

        BF16Frag pa;
        pa.q[0] = *(const uint4*)&Pw[w][ln][8 * g];
        pa.q[1] = *(const uint4*)&Pw[w][ln][16 + 8 * g];

#pragma unroll
        for (int t = 0; t < 4; ++t) {
            BF16Frag b;
            const bf16_t* p = &Vt[t * 16 + ln][8 * g];
            b.q[0] = *(const uint4*)p;
            b.q[1] = *(const uint4*)(p + 16);
            accO[t] = WMMA_BF16(pa.v, b.v, accO[t]);
        }
    }

#pragma unroll
    for (int t = 0; t < 4; ++t)
#pragma unroll
        for (int j = 0; j < 8; ++j) {
            const int row = q0 + j + 8 * g;
            const int col = h * 64 + t * 16 + ln;
            o[(size_t)row * D + col] = f2bf(accO[t][j] / l_run[j]);
        }
}

// --------------------------------------------------------------------------
extern "C" void kernel_launch(void* const* d_in, const int* in_sizes, int n_in,
                              void* d_out, int out_size, void* d_ws, size_t ws_size,
                              hipStream_t stream) {
    constexpr int S = 2048, D = 768, H = 12;
    constexpr int D3 = 3 * D;   // 2304
    constexpr int DH = D / 2;   // 384

    const float* exp_enc  = (const float*)d_in[0];
    const float* pert_enc = (const float*)d_in[1];
    const float* in_w     = (const float*)d_in[2];
    const float* in_b     = (const float*)d_in[3];
    const float* out_w    = (const float*)d_in[4];
    const float* out_b    = (const float*)d_in[5];
    const float* ln0_g    = (const float*)d_in[6];
    const float* ln0_b    = (const float*)d_in[7];
    const float* ln1_g    = (const float*)d_in[8];
    const float* ln1_b    = (const float*)d_in[9];
    const float* ln2_g    = (const float*)d_in[10];
    const float* ln2_b    = (const float*)d_in[11];
    const float* w1       = (const float*)d_in[12];
    const float* b1       = (const float*)d_in[13];
    const float* w2       = (const float*)d_in[14];
    const float* b2       = (const float*)d_in[15];
    const int*   adj      = (const int*)d_in[16];

    char* p = (char*)d_ws;
    auto carve = [&](size_t bytes) {
        char* r = p;
        p += (bytes + 255) & ~(size_t)255;
        return r;
    };
    float*  xin_f   = (float*)carve((size_t)S * D * 4);
    bf16_t* xin_b16 = (bf16_t*)carve((size_t)S * D * 2);
    bf16_t* qkv_b16 = (bf16_t*)carve((size_t)S * D3 * 2);
    bf16_t* attn_o  = (bf16_t*)carve((size_t)S * D * 2);
    float*  tmp_f   = (float*)carve((size_t)S * D * 4);  // out-proj, later FF2
    float*  x_f     = (float*)carve((size_t)S * D * 4);
    bf16_t* x_b16   = (bf16_t*)carve((size_t)S * D * 2);
    bf16_t* h_b16   = (bf16_t*)carve((size_t)S * DH * 2);
    bf16_t* win_b   = (bf16_t*)carve((size_t)D3 * D * 2);
    bf16_t* wout_b  = (bf16_t*)carve((size_t)D * D * 2);
    bf16_t* w1_b    = (bf16_t*)carve((size_t)DH * D * 2);
    bf16_t* w2_b    = (bf16_t*)carve((size_t)D * DH * 2);

    // 1) weight conversion
    k_f32_to_bf16<<<1024, 256, 0, stream>>>(in_w,  win_b,  D3 * D);
    k_f32_to_bf16<<<512,  256, 0, stream>>>(out_w, wout_b, D * D);
    k_f32_to_bf16<<<512,  256, 0, stream>>>(w1,    w1_b,   DH * D);
    k_f32_to_bf16<<<512,  256, 0, stream>>>(w2,    w2_b,   D * DH);

    // 2) x_in = LN0(exp + pert)
    k_add_ln<<<S, 256, 0, stream>>>(exp_enc, pert_enc, ln0_g, ln0_b, xin_f, xin_b16, D);

    // 3) qkv = x_in @ in_proj_w.T + in_proj_b
    k_gemm_wmma<<<dim3(D3 / 128, S / 128), 256, 0, stream>>>(
        xin_b16, win_b, in_b, nullptr, qkv_b16, S, D3, D, 0);

    // 4) masked multi-head attention
    k_attention<<<dim3(S / 128, H), 256, 0, stream>>>(qkv_b16, adj, attn_o, S, D, H);

    // 5) out projection
    k_gemm_wmma<<<dim3(D / 128, S / 128), 256, 0, stream>>>(
        attn_o, wout_b, out_b, tmp_f, nullptr, S, D, D, 0);

    // 6) x = LN1(out_proj + x_in)
    k_add_ln<<<S, 256, 0, stream>>>(tmp_f, xin_f, ln1_g, ln1_b, x_f, x_b16, D);

    // 7) h = relu(x @ w1.T + b1)
    k_gemm_wmma<<<dim3(DH / 128, S / 128), 256, 0, stream>>>(
        x_b16, w1_b, b1, nullptr, h_b16, S, DH, D, 1);

    // 8) ff = h @ w2.T + b2
    k_gemm_wmma<<<dim3(D / 128, S / 128), 256, 0, stream>>>(
        h_b16, w2_b, b2, tmp_f, nullptr, S, D, DH, 0);

    // 9) out = LN2(x + ff)
    k_add_ln<<<S, 256, 0, stream>>>(tmp_f, x_f, ln2_g, ln2_b, (float*)d_out, nullptr, D);

    (void)in_sizes; (void)n_in; (void)out_size; (void)ws_size;
}
